// Brain_25194278159135
// MI455X (gfx1250) — compile-verified, matrix operand order checked by reading, not run on hardware
//
#include <hip/hip_runtime.h>
#include <hip/hip_bf16.h>

// ---------------------------------------------------------------------------
// Model dims (from reference)
// ---------------------------------------------------------------------------
#define B_    32
#define NLOC_ 81
#define SBO_  35
#define SPO_  40
#define E_    224
#define NH_   8
#define DH_   28          // E/NH
#define NLAY_ 10
#define DFF_  224
#define H_    200
#define MSGLOG_ 20
#define MSGEMB_ 100
#define NACT_ 13030
#define NPOW_ 7
#define MAXL_ 17

#define ROWS_  (B_ * NLOC_)          // 2592 (= 81*32)
#define BP_    (B_ * NPOW_)          // 224
#define SEQM_  (MAXL_ * BP_)         // 3808 (= 119*32)
#define G4H_   (4 * H_)              // 800
#define HPAD_  224                   // 200 padded to mult of 32
#define KENC_  (SBO_ + SPO_)         // 75
#define KENCP_ 96
#define KMSG_  (MSGLOG_ * MSGEMB_)   // 2000
#define KMSGP_ 2016
#define KSEQ_  (2 * E_)              // 448
#define KXV_   (NLOC_ * E_ + E_)     // 18368 (mult of 32)

typedef __attribute__((ext_vector_type(16))) _Float16 v16h;
typedef __attribute__((ext_vector_type(8)))  _Float16 v8h;
typedef __attribute__((ext_vector_type(8)))  float    v8f;

// Load one 16-lane WMMA operand fragment for k-chunk starting at p:
// halves 0..7 = K[g*8 .. g*8+7], halves 8..15 = K[16+g*8 .. 16+g*8+7]
// (caller passes p = row + k0 + g*8; second load is +16 halves)
__device__ __forceinline__ v16h ldfrag(const _Float16* __restrict__ p)
{
    v8h lo = *(const v8h*)(p);
    v8h hi = *(const v8h*)(p + 16);
    v16h r;
#pragma unroll
    for (int j = 0; j < 8; ++j) { r[j] = lo[j]; r[8 + j] = hi[j]; }
    return r;
}

// ---------------------------------------------------------------------------
// WMMA GEMM, 2x2 register-blocked (one wave owns a 32x32 output tile):
//   C[m,n] = sum_k A16[m,k] * W16[n,k]            (C = A @ W^T)
//   A16: [M, Kp] f16 row-major, Kp % 32 == 0, M % 32 == 0
//   W16: [N, Kp] f16 row-major (rows clamped when N % 32 != 0)
// Each A/B fragment feeds two v_wmma -> 2 b128 loads per WMMA.
// Epilogue: +bias[n], ReLU, *row_scale[m], optional (s,bp)->(bp,s) permuted
// store for the policy head (m = s*224+bp -> out row bp*17+s).
// ---------------------------------------------------------------------------
__device__ __forceinline__ void store_tile(
    float* __restrict__ C, const v8f& acc, int mbase, int cn, int N,
    const float* __restrict__ bias, const float* __restrict__ row_scale,
    int relu, int permute_sb)
{
    if (cn >= N) return;
    float bv = bias ? bias[cn] : 0.f;
#pragma unroll
    for (int r = 0; r < 8; ++r) {
        int mm = mbase + r;
        float v = acc[r] + bv;
        if (relu) v = v > 0.f ? v : 0.f;
        if (row_scale) v *= row_scale[mm];
        size_t orow = permute_sb ? (size_t)((mm % BP_) * MAXL_ + mm / BP_)
                                 : (size_t)mm;
        C[orow * (size_t)N + cn] = v;
    }
}

__global__ __launch_bounds__(32) void gemm_wmma_f16(
    const _Float16* __restrict__ A, const _Float16* __restrict__ W,
    float* __restrict__ C,
    const float* __restrict__ bias, const float* __restrict__ row_scale,
    int M, int N, int Kp, int relu, int permute_sb)
{
    const int lane = threadIdx.x & 31;
    const int m0 = blockIdx.x << 5;          // 32-row tile
    const int n0 = blockIdx.y << 5;          // 32-col tile
    const int g  = lane >> 4;                // K-half group
    const int hl = lane & 15;

    const _Float16* __restrict__ ap0 = A + (size_t)(m0 + hl) * Kp + g * 8;
    const _Float16* __restrict__ ap1 = A + (size_t)(m0 + 16 + hl) * Kp + g * 8;
    int br0 = n0 + hl;      if (br0 >= N) br0 = N - 1;   // clamp (never stored)
    int br1 = n0 + 16 + hl; if (br1 >= N) br1 = N - 1;
    const _Float16* __restrict__ wp0 = W + (size_t)br0 * Kp + g * 8;
    const _Float16* __restrict__ wp1 = W + (size_t)br1 * Kp + g * 8;

    v8f acc00 = {}, acc01 = {}, acc10 = {}, acc11 = {};
    for (int k0 = 0; k0 < Kp; k0 += 32) {
        if (k0 + 128 < Kp) {                 // pull next chunks toward L0/L2
            __builtin_prefetch(ap0 + k0 + 128, 0, 3);
            __builtin_prefetch(ap1 + k0 + 128, 0, 3);
            __builtin_prefetch(wp0 + k0 + 128, 0, 3);
            __builtin_prefetch(wp1 + k0 + 128, 0, 3);
        }
        v16h a0 = ldfrag(ap0 + k0);
        v16h a1 = ldfrag(ap1 + k0);
        v16h b0 = ldfrag(wp0 + k0);
        v16h b1 = ldfrag(wp1 + k0);
        acc00 = __builtin_amdgcn_wmma_f32_16x16x32_f16(
            false, a0, false, b0, (short)0, acc00, false, false);
        acc01 = __builtin_amdgcn_wmma_f32_16x16x32_f16(
            false, a0, false, b1, (short)0, acc01, false, false);
        acc10 = __builtin_amdgcn_wmma_f32_16x16x32_f16(
            false, a1, false, b0, (short)0, acc10, false, false);
        acc11 = __builtin_amdgcn_wmma_f32_16x16x32_f16(
            false, a1, false, b1, (short)0, acc11, false, false);
    }

    const int cn0 = n0 + hl;
    const int cn1 = n0 + 16 + hl;
    const int mb0 = m0 + 8 * g;
    const int mb1 = m0 + 16 + 8 * g;
    store_tile(C, acc00, mb0, cn0, N, bias, row_scale, relu, permute_sb);
    store_tile(C, acc01, mb0, cn1, N, bias, row_scale, relu, permute_sb);
    store_tile(C, acc10, mb1, cn0, N, bias, row_scale, relu, permute_sb);
    store_tile(C, acc11, mb1, cn1, N, bias, row_scale, relu, permute_sb);
}

// ---------------------------------------------------------------------------
// f32 -> f16 convert with zero pad of K -> Kp
// ---------------------------------------------------------------------------
__global__ void cvt_pad(const float* __restrict__ src, _Float16* __restrict__ dst,
                        int R, int K, int Kp)
{
    size_t total = (size_t)R * Kp;
    for (size_t i = blockIdx.x * (size_t)blockDim.x + threadIdx.x; i < total;
         i += (size_t)gridDim.x * blockDim.x) {
        size_t r = i / (size_t)Kp;
        int k = (int)(i - r * (size_t)Kp);
        dst[i] = (k < K) ? (_Float16)src[r * (size_t)K + k] : (_Float16)0.f;
    }
}

__global__ void zero16(_Float16* p, int n)
{
    for (int i = blockIdx.x * blockDim.x + threadIdx.x; i < n;
         i += gridDim.x * blockDim.x) p[i] = (_Float16)0.f;
}

__global__ void zerof(float* p, int n)
{
    for (int i = blockIdx.x * blockDim.x + threadIdx.x; i < n;
         i += gridDim.x * blockDim.x) p[i] = 0.f;
}

// ---------------------------------------------------------------------------
// Encoder input: concat(x_bo, x_po) -> f16 [2592, 96] zero-padded
// ---------------------------------------------------------------------------
__global__ void build_enc_in(const float* __restrict__ xb,
                             const float* __restrict__ xp,
                             _Float16* __restrict__ dst)
{
    int total = ROWS_ * KENCP_;
    for (int i = blockIdx.x * blockDim.x + threadIdx.x; i < total;
         i += gridDim.x * blockDim.x) {
        int r = i / KENCP_, k = i % KENCP_;
        float v = 0.f;
        if (k < SBO_)       v = xb[r * SBO_ + k];
        else if (k < KENC_) v = xp[r * SPO_ + (k - SBO_)];
        dst[i] = (_Float16)v;
    }
}

__global__ void add_pos(float* __restrict__ x, const float* __restrict__ pos)
{
    int total = ROWS_ * E_;
    for (int i = blockIdx.x * blockDim.x + threadIdx.x; i < total;
         i += gridDim.x * blockDim.x) {
        int e = i % E_;
        int l = (i / E_) % NLOC_;
        x[i] += pos[l * E_ + e];
    }
}

// ---------------------------------------------------------------------------
// Per-(batch, head) attention: softmax(q k^T / sqrt(Dh)) v
// qkv: [2592, 672] (q | k | v), out: [2592, 224]
// ---------------------------------------------------------------------------
__global__ __launch_bounds__(128) void attention_k(
    const float* __restrict__ qkv, float* __restrict__ out)
{
    __shared__ float qs[NLOC_ * DH_];
    __shared__ float ks[NLOC_ * DH_];
    __shared__ float vs[NLOC_ * DH_];
    __shared__ float ss[NLOC_ * NLOC_];

    int bh = blockIdx.x;
    int b = bh >> 3, h = bh & 7;
    const float* base = qkv + (size_t)b * NLOC_ * (3 * E_);

    for (int idx = threadIdx.x; idx < NLOC_ * DH_; idx += blockDim.x) {
        int i = idx / DH_, d = idx % DH_;
        const float* rp = base + (size_t)i * (3 * E_) + h * DH_ + d;
        qs[idx] = rp[0];
        ks[idx] = rp[E_];
        vs[idx] = rp[2 * E_];
    }
    __syncthreads();

    const float scale = 0.1889822365046136f;   // 1/sqrt(28)
    for (int i = threadIdx.x; i < NLOC_; i += blockDim.x) {
        float mx = -3.0e38f;
        for (int j = 0; j < NLOC_; ++j) {
            float d0 = 0.f;
            for (int d = 0; d < DH_; ++d) d0 += qs[i * DH_ + d] * ks[j * DH_ + d];
            d0 *= scale;
            ss[i * NLOC_ + j] = d0;
            mx = d0 > mx ? d0 : mx;
        }
        float sum = 0.f;
        for (int j = 0; j < NLOC_; ++j) {
            float e = expf(ss[i * NLOC_ + j] - mx);
            ss[i * NLOC_ + j] = e;
            sum += e;
        }
        float inv = 1.f / sum;
        for (int d = 0; d < DH_; ++d) {
            float o = 0.f;
            for (int j = 0; j < NLOC_; ++j) o += ss[i * NLOC_ + j] * vs[j * DH_ + d];
            out[((size_t)b * NLOC_ + i) * E_ + h * DH_ + d] = o * inv;
        }
    }
}

// ---------------------------------------------------------------------------
// out = LN(x + o) * g + b  (row-wise over E=224); one block per row
// ---------------------------------------------------------------------------
__global__ __launch_bounds__(256) void ln_res_k(
    const float* __restrict__ x, const float* __restrict__ o,
    const float* __restrict__ gg, const float* __restrict__ bb,
    float* __restrict__ out)
{
    __shared__ float sh[256];
    int row = blockIdx.x, t = threadIdx.x;
    float v = 0.f;
    if (t < E_) v = x[(size_t)row * E_ + t] + o[(size_t)row * E_ + t];
    sh[t] = v;
    __syncthreads();
    for (int s = 128; s > 0; s >>= 1) { if (t < s) sh[t] += sh[t + s]; __syncthreads(); }
    float mean = sh[0] * (1.f / E_);
    __syncthreads();
    float d = (t < E_) ? v - mean : 0.f;
    sh[t] = d * d;
    __syncthreads();
    for (int s = 128; s > 0; s >>= 1) { if (t < s) sh[t] += sh[t + s]; __syncthreads(); }
    float var = sh[0] * (1.f / E_);
    if (t < E_) out[(size_t)row * E_ + t] = d * rsqrtf(var + 1e-5f) * gg[t] + bb[t];
}

// ---------------------------------------------------------------------------
// LSTM input seq (layer 0): f16 [3808, 448]; m = s*224 + bp
// cols 0..223  = x[b, locs_ix[b,p,s], :], cols 224..447 = msg_emb[b, :]
// ---------------------------------------------------------------------------
__global__ void build_seq(const float* __restrict__ x,
                          const float* __restrict__ msg,
                          const int* __restrict__ locs_ix,
                          _Float16* __restrict__ seq)
{
    int total = SEQM_ * KSEQ_;
    for (int i = blockIdx.x * blockDim.x + threadIdx.x; i < total;
         i += gridDim.x * blockDim.x) {
        int m = i / KSEQ_, k = i % KSEQ_;
        int s = m / BP_, bp = m % BP_;
        int b = bp / NPOW_;
        float v;
        if (k < E_) {
            int loc = locs_ix[bp * MAXL_ + s];
            v = x[((size_t)b * NLOC_ + loc) * E_ + k];
        } else {
            v = msg[b * E_ + (k - E_)];
        }
        seq[i] = (_Float16)v;
    }
}

// ---------------------------------------------------------------------------
// LSTM pointwise gates for one timestep; also writes f16 h for next GEMMs
// Xg: [3808, 800] precomputed x@Wih^T; Hg: [224, 800] = h_{t-1}@Whh^T
// ---------------------------------------------------------------------------
__global__ void lstm_gate(const float* __restrict__ Xg,
                          const float* __restrict__ Hg,
                          const float* __restrict__ bih,
                          const float* __restrict__ bhh,
                          float* __restrict__ c,
                          _Float16* __restrict__ h_cur,   // [224, 224]
                          _Float16* __restrict__ h_all,   // [3808, 224]
                          int t)
{
    int idx = blockIdx.x * blockDim.x + threadIdx.x;
    if (idx >= BP_ * H_) return;
    int row = idx / H_, j = idx % H_;
    const float* xg = Xg + ((size_t)t * BP_ + row) * G4H_;
    const float* hg = Hg + (size_t)row * G4H_;

    float gi = xg[j]         + hg[j]         + bih[j]         + bhh[j];
    float gf = xg[H_ + j]    + hg[H_ + j]    + bih[H_ + j]    + bhh[H_ + j];
    float gg = xg[2*H_ + j]  + hg[2*H_ + j]  + bih[2*H_ + j]  + bhh[2*H_ + j];
    float go = xg[3*H_ + j]  + hg[3*H_ + j]  + bih[3*H_ + j]  + bhh[3*H_ + j];

    float ip = 1.f / (1.f + expf(-gi));
    float fp = 1.f / (1.f + expf(-gf));
    float op = 1.f / (1.f + expf(-go));
    float cn = fp * c[idx] + ip * tanhf(gg);
    c[idx] = cn;
    float hn = op * tanhf(cn);
    h_cur[(size_t)row * HPAD_ + j] = (_Float16)hn;
    h_all[((size_t)t * BP_ + row) * HPAD_ + j] = (_Float16)hn;
}

// ---------------------------------------------------------------------------
// Ragged mask as per-row scale for the policy GEMM: m = s*224 + bp
// ---------------------------------------------------------------------------
__global__ void build_row_scale(const int* __restrict__ locs_len,
                                float* __restrict__ rs)
{
    int m = blockIdx.x * blockDim.x + threadIdx.x;
    if (m >= SEQM_) return;
    int s = m / BP_, bp = m % BP_;
    rs[m] = (s < locs_len[bp]) ? 1.f : 0.f;
}

// ---------------------------------------------------------------------------
// Value-head input: f16 [32, 18368] = concat(x.flat, msg_emb)
// ---------------------------------------------------------------------------
__global__ void build_xv(const float* __restrict__ x,
                         const float* __restrict__ msg,
                         _Float16* __restrict__ xv)
{
    int total = B_ * KXV_;
    for (int i = blockIdx.x * blockDim.x + threadIdx.x; i < total;
         i += gridDim.x * blockDim.x) {
        int b = i / KXV_, k = i % KXV_;
        float v = (k < NLOC_ * E_) ? x[(size_t)b * (NLOC_ * E_) + k]
                                   : msg[b * E_ + (k - NLOC_ * E_)];
        xv[i] = (_Float16)v;
    }
}

// value[b,p] = relu(val1)[b,:] @ lin2_W[p,:] + lin2_b[p]   (val1 already relu'd)
__global__ void value_head2(const float* __restrict__ val1,
                            const float* __restrict__ lin2_W,
                            const float* __restrict__ lin2_b,
                            float* __restrict__ out)
{
    int i = blockIdx.x * blockDim.x + threadIdx.x;
    if (i >= B_ * NPOW_) return;
    int b = i / NPOW_, p = i % NPOW_;
    float s = lin2_b[p];
    for (int e = 0; e < E_; ++e) s += val1[b * E_ + e] * lin2_W[p * E_ + e];
    out[i] = s;
}

// ---------------------------------------------------------------------------
// Host-side orchestration
// ---------------------------------------------------------------------------
static inline int gblocks(long long total, int tpb)
{
    long long b = (total + tpb - 1) / tpb;
    if (b > 4096) b = 4096;
    return (int)b;
}

extern "C" void kernel_launch(void* const* d_in, const int* in_sizes, int n_in,
                              void* d_out, int out_size, void* d_ws, size_t ws_size,
                              hipStream_t stream)
{
    (void)in_sizes; (void)n_in; (void)out_size; (void)ws_size;

    const float* x_bo    = (const float*)d_in[0];
    const float* x_po    = (const float*)d_in[1];
    const float* msg_log = (const float*)d_in[2];
    const float* enc_W   = (const float*)d_in[3];
    const float* enc_b   = (const float*)d_in[4];
    const float* pos_bias= (const float*)d_in[5];
    const float* Wqkv    = (const float*)d_in[6];
    const float* bqkv    = (const float*)d_in[7];
    const float* Wo      = (const float*)d_in[8];
    const float* bo      = (const float*)d_in[9];
    const float* ln1_g   = (const float*)d_in[10];
    const float* ln1_b   = (const float*)d_in[11];
    const float* W1      = (const float*)d_in[12];
    const float* b1      = (const float*)d_in[13];
    const float* W2      = (const float*)d_in[14];
    const float* b2      = (const float*)d_in[15];
    const float* ln2_g   = (const float*)d_in[16];
    const float* ln2_b   = (const float*)d_in[17];
    const float* msg_W   = (const float*)d_in[18];
    const float* msg_b   = (const float*)d_in[19];
    const float* Wih0    = (const float*)d_in[20];
    const float* Whh0    = (const float*)d_in[21];
    const float* bih0    = (const float*)d_in[22];
    const float* bhh0    = (const float*)d_in[23];
    const float* Wih1    = (const float*)d_in[24];
    const float* Whh1    = (const float*)d_in[25];
    const float* bih1    = (const float*)d_in[26];
    const float* bhh1    = (const float*)d_in[27];
    const float* pol_W   = (const float*)d_in[28];
    const float* pol_b   = (const float*)d_in[29];
    const float* lin1_W  = (const float*)d_in[30];
    const float* lin1_b  = (const float*)d_in[31];
    const float* lin2_W  = (const float*)d_in[32];
    const float* lin2_b  = (const float*)d_in[33];
    const int*   locs_ix = (const int*)d_in[34];
    const int*   locs_len= (const int*)d_in[35];

    float* dist    = (float*)d_out;                          // [B,P,S,NACT]
    float* val_out = dist + (size_t)SEQM_ * NACT_;           // [B,P]

    // ---- workspace layout (deterministic cursor) ----
    char* cur = (char*)d_ws;
    auto A16 = [&](size_t n) {
        _Float16* r = (_Float16*)cur;
        cur += ((n * 2 + 255) & ~(size_t)255);
        return r;
    };
    auto A32 = [&](size_t n) {
        float* r = (float*)cur;
        cur += ((n * 4 + 255) & ~(size_t)255);
        return r;
    };

    // f16 weights (converted every call; no cached state)
    _Float16* w_enc  = A16((size_t)E_ * KENCP_);
    _Float16* w_qkv  = A16((size_t)NLAY_ * 3 * E_ * E_);
    _Float16* w_o    = A16((size_t)NLAY_ * E_ * E_);
    _Float16* w_1    = A16((size_t)NLAY_ * DFF_ * E_);
    _Float16* w_2    = A16((size_t)NLAY_ * E_ * DFF_);
    _Float16* w_msg  = A16((size_t)E_ * KMSGP_);
    _Float16* w_ih0  = A16((size_t)G4H_ * KSEQ_);
    _Float16* w_hh0  = A16((size_t)G4H_ * HPAD_);
    _Float16* w_ih1  = A16((size_t)G4H_ * HPAD_);
    _Float16* w_hh1  = A16((size_t)G4H_ * HPAD_);
    _Float16* w_pol  = A16((size_t)NACT_ * HPAD_);
    _Float16* w_lin1 = A16((size_t)E_ * KXV_);

    // activations
    _Float16* enc_in  = A16((size_t)ROWS_ * KENCP_);
    _Float16* act16   = A16((size_t)ROWS_ * E_);
    float*    x       = A32((size_t)ROWS_ * E_);
    float*    qkv     = A32((size_t)ROWS_ * 3 * E_);
    float*    attnout = A32((size_t)ROWS_ * E_);     // reused as ff1 output
    float*    tmp     = A32((size_t)ROWS_ * E_);
    _Float16* msg16   = A16((size_t)B_ * KMSGP_);
    float*    msg_emb = A32((size_t)B_ * E_);
    _Float16* seq16   = A16((size_t)SEQM_ * KSEQ_);
    float*    Xg      = A32((size_t)SEQM_ * G4H_);
    float*    Hg      = A32((size_t)BP_ * G4H_);
    _Float16* hcur16  = A16((size_t)BP_ * HPAD_);
    float*    cbuf    = A32((size_t)BP_ * H_);
    _Float16* hall0   = A16((size_t)SEQM_ * HPAD_);
    _Float16* hall1   = A16((size_t)SEQM_ * HPAD_);
    float*    rscale  = A32((size_t)SEQM_);
    _Float16* xv16    = A16((size_t)B_ * KXV_);
    float*    val1    = A32((size_t)B_ * E_);

    auto cvt = [&](const float* s, _Float16* d, int R, int K, int Kp) {
        cvt_pad<<<gblocks((long long)R * Kp, 256), 256, 0, stream>>>(s, d, R, K, Kp);
    };
    auto gemm = [&](const _Float16* A, const _Float16* W, float* C,
                    const float* bias, const float* rs,
                    int M, int N, int Kp, int relu, int perm) {
        dim3 grid(M / 32, (N + 31) / 32);
        gemm_wmma_f16<<<grid, 32, 0, stream>>>(A, W, C, bias, rs, M, N, Kp, relu, perm);
    };

    // ---- weight conversion (f32 -> f16, K zero-padded) ----
    cvt(enc_W,  w_enc,  E_,           KENC_, KENCP_);
    cvt(Wqkv,   w_qkv,  NLAY_ * 3*E_, E_,    E_);
    cvt(Wo,     w_o,    NLAY_ * E_,   E_,    E_);
    cvt(W1,     w_1,    NLAY_ * DFF_, E_,    E_);
    cvt(W2,     w_2,    NLAY_ * E_,   DFF_,  DFF_);
    cvt(msg_W,  w_msg,  E_,           KMSG_, KMSGP_);
    cvt(Wih0,   w_ih0,  G4H_,         KSEQ_, KSEQ_);
    cvt(Whh0,   w_hh0,  G4H_,         H_,    HPAD_);
    cvt(Wih1,   w_ih1,  G4H_,         H_,    HPAD_);
    cvt(Whh1,   w_hh1,  G4H_,         H_,    HPAD_);
    cvt(pol_W,  w_pol,  NACT_,        H_,    HPAD_);
    cvt(lin1_W, w_lin1, E_,           KXV_,  KXV_);

    // ---- encoder linear + positional bias ----
    build_enc_in<<<gblocks(ROWS_ * KENCP_, 256), 256, 0, stream>>>(x_bo, x_po, enc_in);
    gemm(enc_in, w_enc, x, enc_b, nullptr, ROWS_, E_, KENCP_, 0, 0);
    add_pos<<<gblocks(ROWS_ * E_, 256), 256, 0, stream>>>(x, pos_bias);

    // ---- transformer layers ----
    for (int l = 0; l < NLAY_; ++l) {
        cvt(x, act16, ROWS_, E_, E_);
        gemm(act16, w_qkv + (size_t)l * 3 * E_ * E_, qkv,
             bqkv + l * 3 * E_, nullptr, ROWS_, 3 * E_, E_, 0, 0);
        attention_k<<<B_ * NH_, 128, 0, stream>>>(qkv, attnout);
        cvt(attnout, act16, ROWS_, E_, E_);
        gemm(act16, w_o + (size_t)l * E_ * E_, tmp,
             bo + l * E_, nullptr, ROWS_, E_, E_, 0, 0);
        ln_res_k<<<ROWS_, 256, 0, stream>>>(x, tmp, ln1_g + l * E_, ln1_b + l * E_, x);
        cvt(x, act16, ROWS_, E_, E_);
        gemm(act16, w_1 + (size_t)l * DFF_ * E_, attnout,
             b1 + l * DFF_, nullptr, ROWS_, DFF_, E_, 1, 0);
        cvt(attnout, act16, ROWS_, DFF_, DFF_);
        gemm(act16, w_2 + (size_t)l * E_ * DFF_, tmp,
             b2 + l * E_, nullptr, ROWS_, E_, DFF_, 0, 0);
        ln_res_k<<<ROWS_, 256, 0, stream>>>(x, tmp, ln2_g + l * E_, ln2_b + l * E_, x);
    }

    // ---- message embedding ----
    cvt(msg_log, msg16, B_, KMSG_, KMSGP_);
    gemm(msg16, w_msg, msg_emb, msg_b, nullptr, B_, E_, KMSGP_, 1, 0);

    // ---- LSTM input sequence + layer 0 ----
    build_seq<<<gblocks((long long)SEQM_ * KSEQ_, 256), 256, 0, stream>>>(
        x, msg_emb, locs_ix, seq16);
    gemm(seq16, w_ih0, Xg, nullptr, nullptr, SEQM_, G4H_, KSEQ_, 0, 0);
    zero16<<<gblocks(BP_ * HPAD_, 256), 256, 0, stream>>>(hcur16, BP_ * HPAD_);
    zero16<<<gblocks(SEQM_ * HPAD_, 256), 256, 0, stream>>>(hall0, SEQM_ * HPAD_);
    zerof<<<gblocks(BP_ * H_, 256), 256, 0, stream>>>(cbuf, BP_ * H_);
    for (int t = 0; t < MAXL_; ++t) {
        gemm(hcur16, w_hh0, Hg, nullptr, nullptr, BP_, G4H_, HPAD_, 0, 0);
        lstm_gate<<<(BP_ * H_ + 255) / 256, 256, 0, stream>>>(
            Xg, Hg, bih0, bhh0, cbuf, hcur16, hall0, t);
    }

    // ---- LSTM layer 1 ----
    gemm(hall0, w_ih1, Xg, nullptr, nullptr, SEQM_, G4H_, HPAD_, 0, 0);
    zero16<<<gblocks(BP_ * HPAD_, 256), 256, 0, stream>>>(hcur16, BP_ * HPAD_);
    zero16<<<gblocks(SEQM_ * HPAD_, 256), 256, 0, stream>>>(hall1, SEQM_ * HPAD_);
    zerof<<<gblocks(BP_ * H_, 256), 256, 0, stream>>>(cbuf, BP_ * H_);
    for (int t = 0; t < MAXL_; ++t) {
        gemm(hcur16, w_hh1, Hg, nullptr, nullptr, BP_, G4H_, HPAD_, 0, 0);
        lstm_gate<<<(BP_ * H_ + 255) / 256, 256, 0, stream>>>(
            Xg, Hg, bih1, bhh1, cbuf, hcur16, hall1, t);
    }

    // ---- policy head: dist written directly in [B,P,S,NACT] order, masked ----
    build_row_scale<<<(SEQM_ + 255) / 256, 256, 0, stream>>>(locs_len, rscale);
    gemm(hall1, w_pol, dist, pol_b, rscale, SEQM_, NACT_, HPAD_, 0, 1);

    // ---- value head ----
    build_xv<<<gblocks((long long)B_ * KXV_, 256), 256, 0, stream>>>(x, msg_emb, xv16);
    gemm(xv16, w_lin1, val1, lin1_b, nullptr, B_, E_, KXV_, 1, 0);
    value_head2<<<1, 256, 0, stream>>>(val1, lin2_W, lin2_b, val_out);
}
